// MyGRUCell_55843164783146
// MI455X (gfx1250) — matrix operand order repaired and checked by reading, not compiled, not run on hardware
//
#include <hip/hip_runtime.h>
#include <hip/hip_bf16.h>
#include <stdint.h>

// ---------------------------------------------------------------------------
// GRU forward for MI455X (gfx1250): bf16 WMMA GEMMs, fp32 gates in registers.
// NB=6400 rows -> 200 blocks x 32 rows (2 WMMA M-tiles).
// Wave w owns H-columns [16w,16w+16): accumulates its z / r / hh slices
// (N-tiles w, w+16, w+32) in registers; gates + h-update fully in-register.
// Weight B-fragments stream from L2 each timestep; an anti-LICM asm barrier
// on a zero scalar offset keeps the loads in-loop *and* in the global
// address space (global_load_b128 saddr form, not flat).
// ---------------------------------------------------------------------------

typedef __attribute__((ext_vector_type(16))) __bf16 v16bf;
typedef __attribute__((ext_vector_type(8)))  float  v8f;

#define T_STEPS 64
#define D_IN    128
#define H_DIM   256
#define G3      768           // 3*H
#define NB_TOT  6400          // N*B
#define MT      2             // M-tiles (16 rows each) per block
#define ROWS    (MT * 16)     // 32
#define WAVES   16
#define THREADS (WAVES * 32)
#define KT_X    (D_IN  / 32)  // 4
#define KT_H    (H_DIM / 32)  // 8
#define NTILES  (G3 / 16)     // 48

__device__ __forceinline__ unsigned short f32_bf16(float f) {
  unsigned int u = __float_as_uint(f);
  u += 0x7FFFu + ((u >> 16) & 1u);           // round-to-nearest-even
  return (unsigned short)(u >> 16);
}

union FragU { uint4 u4[2]; v16bf v; };

// B 32x16 bf16 fragment, packed: lane L -> col N=L&15, K=(L>=16?16:0)+e,
// 16 bf16 per lane contiguous (32B).
__device__ __forceinline__ v16bf load_bfrag(const unsigned short* __restrict__ base,
                                            int kt, int nt, int lane) {
  const unsigned short* p = base + (size_t)(kt * NTILES + nt) * 512 + lane * 16;
  FragU fu;
  fu.u4[0] = *(const uint4*)p;
  fu.u4[1] = *(const uint4*)(p + 8);
  return fu.v;
}

// A 16x32 bf16 fragment from an LDS row (row-major bf16 tile).
__device__ __forceinline__ v16bf load_afrag(const unsigned short* rowptr, int k0) {
  FragU fu;
  fu.u4[0] = *(const uint4*)(rowptr + k0);
  fu.u4[1] = *(const uint4*)(rowptr + k0 + 8);
  return fu.v;
}

// ---------------------------------------------------------------------------
__global__ __launch_bounds__(32) void pack_weights(
    const float* __restrict__ Wx, const float* __restrict__ Wh,
    unsigned short* __restrict__ wxp, unsigned short* __restrict__ whp) {
  int f = blockIdx.x;
  int lane = threadIdx.x;
  const float* src; unsigned short* dst; int g;
  if (f < KT_X * NTILES) { src = Wx; dst = wxp; g = f; }
  else                   { src = Wh; dst = whp; g = f - KT_X * NTILES; }
  int kt = g / NTILES, nt = g % NTILES;
  int kbase = kt * 32 + ((lane >> 4) & 1) * 16;
  int col   = nt * 16 + (lane & 15);
  unsigned short tmp[16];
#pragma unroll
  for (int e = 0; e < 16; ++e)
    tmp[e] = f32_bf16(src[(size_t)(kbase + e) * G3 + col]);
  unsigned short* o = dst + (size_t)g * 512 + lane * 16;
  *(uint4*)(o)     = *(const uint4*)(tmp);
  *(uint4*)(o + 8) = *(const uint4*)(tmp + 8);
}

// ---------------------------------------------------------------------------
__global__ __launch_bounds__(THREADS) void gru_kernel(
    const float* __restrict__ xin,    // (NB, T, D)
    const float* __restrict__ h0,     // (NB, H)
    const float* __restrict__ b_in,   // (G3)
    const float* __restrict__ b_rec,  // (G3)
    const unsigned short* __restrict__ wxp_in,
    const unsigned short* __restrict__ whp_in,
    float* __restrict__ out,          // (NB, T, H)
    float* __restrict__ state)        // (NB, H)
{
  __shared__ unsigned short s_x[ROWS][D_IN];   //  8 KB : x tile, bf16
  __shared__ unsigned short s_h[ROWS][H_DIM];  // 16 KB : h, bf16 (GEMM A side)

  const int tid   = threadIdx.x;
  const int lane  = tid & 31;
  const int wave  = tid >> 5;
  const int row0  = blockIdx.x * ROWS;
  const int am    = lane & 15;                 // A row within M-tile / D col
  const int ak    = ((lane >> 4) & 1) * 16;    // A K sub-offset by lane half
  const int half8 = ((lane >> 4) & 1) * 8;     // C/D row base by lane half
  const int col   = wave * 16 + am;            // this lane's H column (0..255)

  // per-lane gate biases (fixed across timesteps)
  const float bz  = b_in[col]       + b_rec[col];
  const float brr = b_in[col + 256] + b_rec[col + 256];
  const float bhx = b_in[col + 512];
  const float bhr = b_rec[col + 512];

  // h carried in registers (fp32), mirrored in LDS as bf16 for the GEMM A side
  float hprev[MT][8];
#pragma unroll
  for (int m = 0; m < MT; ++m)
#pragma unroll
    for (int r = 0; r < 8; ++r)
      hprev[m][r] = h0[(size_t)(row0 + m * 16 + half8 + r) * H_DIM + col];

  for (int i = tid; i < ROWS * H_DIM; i += THREADS) {
    int mrow = i >> 8, c = i & (H_DIM - 1);
    s_h[mrow][c] = f32_bf16(h0[(size_t)(row0 + mrow) * H_DIM + c]);
  }
  __syncthreads();

  const int xr = tid >> 4;          // 0..31 rows
  const int xc = (tid & 15) * 8;    // 8 floats per thread

  for (int t = 0; t < T_STEPS; ++t) {
    // Anti-LICM: opaque zero offset, redefined every iteration by a volatile
    // asm, keeps the B-fragment loads inside the loop (L2 streaming) while
    // preserving global address-space provenance of the weight pointers.
    unsigned int zoff = 0;
    asm volatile("" : "+s"(zoff));
    const unsigned short* __restrict__ wxp = wxp_in + zoff;
    const unsigned short* __restrict__ whp = whp_in + zoff;

    // ---- stage x tile fp32 -> bf16 -> LDS; prefetch next step ----
    {
      const float* gx = xin + ((size_t)(row0 + xr) * T_STEPS + t) * D_IN + xc;
      float4 v0 = *(const float4*)gx;
      float4 v1 = *(const float4*)(gx + 4);
      uint4 p;
      p.x = (unsigned int)f32_bf16(v0.x) | ((unsigned int)f32_bf16(v0.y) << 16);
      p.y = (unsigned int)f32_bf16(v0.z) | ((unsigned int)f32_bf16(v0.w) << 16);
      p.z = (unsigned int)f32_bf16(v1.x) | ((unsigned int)f32_bf16(v1.y) << 16);
      p.w = (unsigned int)f32_bf16(v1.z) | ((unsigned int)f32_bf16(v1.w) << 16);
      *(uint4*)&s_x[xr][xc] = p;
      if (t + 1 < T_STEPS) __builtin_prefetch(gx + D_IN, 0, 0);
    }
    __syncthreads();

    // ---- accumulators: z / r combine x+h parts; hh keeps them separate ----
    v8f accz[MT], accr[MT], acchx[MT], acchh[MT];
#pragma unroll
    for (int m = 0; m < MT; ++m) {
      accz[m]  = (v8f){0.f, 0.f, 0.f, 0.f, 0.f, 0.f, 0.f, 0.f};
      accr[m]  = accz[m];
      acchx[m] = accz[m];
      acchh[m] = accz[m];
    }

    // ---- x GEMM part: K = 128 ----
#pragma unroll
    for (int kt = 0; kt < KT_X; ++kt) {
      v16bf A[MT];
#pragma unroll
      for (int m = 0; m < MT; ++m)
        A[m] = load_afrag(&s_x[m * 16 + am][0], kt * 32 + ak);
      v16bf Bz = load_bfrag(wxp, kt, wave,      lane);
      v16bf Br = load_bfrag(wxp, kt, wave + 16, lane);
      v16bf Bh = load_bfrag(wxp, kt, wave + 32, lane);
#pragma unroll
      for (int m = 0; m < MT; ++m) {
        accz[m]  = __builtin_amdgcn_wmma_f32_16x16x32_bf16(false, A[m], false, Bz, (short)0, accz[m],  false, false);
        accr[m]  = __builtin_amdgcn_wmma_f32_16x16x32_bf16(false, A[m], false, Br, (short)0, accr[m],  false, false);
        acchx[m] = __builtin_amdgcn_wmma_f32_16x16x32_bf16(false, A[m], false, Bh, (short)0, acchx[m], false, false);
      }
    }

    // ---- h GEMM part: K = 256 ----
#pragma unroll
    for (int kt = 0; kt < KT_H; ++kt) {
      v16bf A[MT];
#pragma unroll
      for (int m = 0; m < MT; ++m)
        A[m] = load_afrag(&s_h[m * 16 + am][0], kt * 32 + ak);
      v16bf Bz = load_bfrag(whp, kt, wave,      lane);
      v16bf Br = load_bfrag(whp, kt, wave + 16, lane);
      v16bf Bh = load_bfrag(whp, kt, wave + 32, lane);
#pragma unroll
      for (int m = 0; m < MT; ++m) {
        accz[m]  = __builtin_amdgcn_wmma_f32_16x16x32_bf16(false, A[m], false, Bz, (short)0, accz[m],  false, false);
        accr[m]  = __builtin_amdgcn_wmma_f32_16x16x32_bf16(false, A[m], false, Br, (short)0, accr[m],  false, false);
        acchh[m] = __builtin_amdgcn_wmma_f32_16x16x32_bf16(false, A[m], false, Bh, (short)0, acchh[m], false, false);
      }
    }
    __syncthreads();   // all GEMM reads of s_h complete before updating it

    // ---- gates + h update, fully in registers ----
#pragma unroll
    for (int m = 0; m < MT; ++m) {
#pragma unroll
      for (int r = 0; r < 8; ++r) {
        const int row = m * 16 + half8 + r;
        float z  = 1.f / (1.f + __expf(-(accz[m][r] + bz)));
        float rr = 1.f / (1.f + __expf(-(accr[m][r] + brr)));
        float hh = tanhf(acchx[m][r] + bhx + rr * (acchh[m][r] + bhr));
        float hn = z * hprev[m][r] + (1.f - z) * hh;
        hprev[m][r] = hn;
        s_h[row][col] = f32_bf16(hn);
        out[((size_t)(row0 + row) * T_STEPS + t) * H_DIM + col] = hn;
      }
    }
    // next-iteration barrier (after x staging) orders s_h writes vs GEMM reads
  }

  // ---- final state ----
#pragma unroll
  for (int m = 0; m < MT; ++m)
#pragma unroll
    for (int r = 0; r < 8; ++r)
      state[(size_t)(row0 + m * 16 + half8 + r) * H_DIM + col] = hprev[m][r];
}

// ---------------------------------------------------------------------------
extern "C" void kernel_launch(void* const* d_in, const int* in_sizes, int n_in,
                              void* d_out, int out_size, void* d_ws, size_t ws_size,
                              hipStream_t stream) {
  (void)in_sizes; (void)n_in; (void)out_size; (void)ws_size;
  const float* inputs = (const float*)d_in[1];   // (N,B,T,D)
  const float* states = (const float*)d_in[2];   // (N,B,H)
  const float* Wx     = (const float*)d_in[3];   // (D,3H)
  const float* Wh     = (const float*)d_in[4];   // (H,3H)
  const float* b_in   = (const float*)d_in[5];   // (3H)
  const float* b_rec  = (const float*)d_in[6];   // (3H)

  float* out   = (float*)d_out;                           // (N,B,T,H)
  float* state = out + (size_t)NB_TOT * T_STEPS * H_DIM;  // (N,B,H)

  unsigned short* wxp = (unsigned short*)d_ws;             // 192 KB packed bf16
  unsigned short* whp = wxp + (size_t)KT_X * NTILES * 512; // 384 KB packed bf16

  pack_weights<<<(KT_X + KT_H) * NTILES, 32, 0, stream>>>(Wx, Wh, wxp, whp);
  gru_kernel<<<NB_TOT / ROWS, THREADS, 0, stream>>>(
      inputs, states, b_in, b_rec, wxp, whp, out, state);
}
